// GCNEncoder_60155311947910
// MI455X (gfx1250) — matrix operand order repaired and checked by reading, not compiled, not run on hardware
//
#include <hip/hip_runtime.h>

constexpr int Bc = 32, Tc = 512, Dc = 256;

typedef __attribute__((ext_vector_type(16))) __bf16 v16bf;
typedef __attribute__((ext_vector_type(8)))  float  v8f;

#define DEVFN static __device__ __forceinline__

DEVFN __bf16 f2bf(float f) {
  unsigned u = __builtin_bit_cast(unsigned, f);
  u += 0x7FFFu + ((u >> 16) & 1u);                 // round-to-nearest-even
  unsigned short h = (unsigned short)(u >> 16);
  return __builtin_bit_cast(__bf16, h);
}
DEVFN float bf2f(__bf16 x) {
  unsigned short s = __builtin_bit_cast(unsigned short, x);
  unsigned u = (unsigned)s << 16;
  return __builtin_bit_cast(float, u);
}

// A-operand tile fragment: 8 bf16 at +0 and 8 bf16 at +16 (elements)
DEVFN v16bf load_tileA(const __bf16* p) {
  v16bf v;
  ((uint4*)&v)[0] = *(const uint4*)(p);
  ((uint4*)&v)[1] = *(const uint4*)(p + 16);
  return v;
}
// B-operand tile fragment: 8 bf16 at +0 and 8 bf16 at +16 (column of B = row of N-major store)
DEVFN v16bf load_tileB(const __bf16* p) {
  v16bf v;
  ((uint4*)&v)[0] = *(const uint4*)(p);
  ((uint4*)&v)[1] = *(const uint4*)(p + 16);
  return v;
}

DEVFN v8f wmma_bf16(v16bf a, v16bf b, v8f c) {
  return __builtin_amdgcn_wmma_f32_16x16x32_bf16(false, a, false, b,
                                                 (short)0, c, false, false);
}

// ---------------------------------------------------------------------------
// Prep: adjacency (int 0/1 -> binarized, or f32 passthrough) -> bf16 row-major
// plus per-row denom = rowsum + 1  (f32, matches reference denom exactly)
// grid (T, B), block 256
// ---------------------------------------------------------------------------
__global__ void __launch_bounds__(256)
gcnenc_adj_prep_int(const int* __restrict__ src, __bf16* __restrict__ adjbf,
                    float* __restrict__ denom) {
  const int t = blockIdx.x, b = blockIdx.y;
  const size_t row = ((size_t)b * Tc + t) * Tc;
  float sum = 0.0f;
  for (int j = threadIdx.x; j < Tc; j += 256) {
    float v = (src[row + j] != 0) ? 1.0f : 0.0f;
    adjbf[row + j] = f2bf(v);
    sum += v;
  }
  __shared__ float red[256];
  red[threadIdx.x] = sum; __syncthreads();
  for (int s = 128; s > 0; s >>= 1) {
    if (threadIdx.x < s) red[threadIdx.x] += red[threadIdx.x + s];
    __syncthreads();
  }
  if (threadIdx.x == 0) denom[(size_t)b * Tc + t] = red[0] + 1.0f;
}

__global__ void __launch_bounds__(256)
gcnenc_adj_prep_f32(const float* __restrict__ src, __bf16* __restrict__ adjbf,
                    float* __restrict__ denom) {
  const int t = blockIdx.x, b = blockIdx.y;
  const size_t row = ((size_t)b * Tc + t) * Tc;
  float sum = 0.0f;
  for (int j = threadIdx.x; j < Tc; j += 256) {
    float v = src[row + j];
    adjbf[row + j] = f2bf(v);
    sum += v;
  }
  __shared__ float red[256];
  red[threadIdx.x] = sum; __syncthreads();
  for (int s = 128; s > 0; s >>= 1) {
    if (threadIdx.x < s) red[threadIdx.x] += red[threadIdx.x + s];
    __syncthreads();
  }
  if (threadIdx.x == 0) denom[(size_t)b * Tc + t] = red[0] + 1.0f;
}

// inputs (B,T,D) f32 -> x^T bf16 [B][D][T]
__global__ void __launch_bounds__(256)
gcnenc_x0_prep(const float* __restrict__ x, __bf16* __restrict__ xT) {
  size_t i = (size_t)blockIdx.x * 256 + threadIdx.x;
  int b = (int)(i / (Tc * Dc));
  int r = (int)(i % (Tc * Dc));
  int t = r / Dc, d = r % Dc;
  xT[((size_t)b * Dc + d) * Tc + t] = f2bf(x[i]);
}

// W (D,D) f32 -> bf16 row-major (already the N-major layout for x@W.T);
// bias2 = 2*b
__global__ void __launch_bounds__(256)
gcnenc_w_prep(const float* __restrict__ W, const float* __restrict__ bsrc,
              __bf16* __restrict__ Wbf, float* __restrict__ bias2) {
  int i = blockIdx.x * 256 + threadIdx.x;
  Wbf[i] = f2bf(W[i]);
  if (i < Dc) bias2[i] = 2.0f * bsrc[i];
}

// ---------------------------------------------------------------------------
// GEMM1:  S[t][d] = sum_j adj[t][j] * x[j][d]  + x[t][d]
//   A = adj bf16 row-major [T][T];  B(N-major) = x^T [D][T];  C -> S [T][D] bf16
//   block 128 (4 waves), wave -> 32(M) x 64(N); block covers 32 x 256.
//   K loop software-pipelined; sched_barrier(0) pins "issue next step's loads"
//   before "current step's WMMAs" so the wait ahead of the WMMA group covers
//   loads issued a full iteration earlier (non-zero loadcnt wait).
// ---------------------------------------------------------------------------
__global__ void __launch_bounds__(128)
gcnenc_gemm1(const __bf16* __restrict__ adj, const __bf16* __restrict__ xT,
             __bf16* __restrict__ S) {
  const int b  = blockIdx.y;
  const int m0 = blockIdx.x * 32;
  const int wv = threadIdx.x >> 5;
  const int ln = threadIdx.x & 31;
  const int n0 = wv * 64;

  const __bf16* A  = adj + (size_t)b * Tc * Tc;
  const __bf16* Bt = xT  + (size_t)b * Dc * Tc;

  const int mA  = ln & 15;           // A: row within tile
  const int kgA = (ln >> 4) * 8;     // A: K group base (0 or 8), +16 for hi half
  const int nB  = ln & 15;           // B: column within tile
  const int khB = (ln >> 4) * 8;     // B: K group base (0 or 8), +16 for hi half

  const __bf16* aRow0 = A + (size_t)(m0 + mA) * Tc + kgA;
  const __bf16* aRow1 = A + (size_t)(m0 + 16 + mA) * Tc + kgA;
  const __bf16* bRow0 = Bt + (size_t)(n0 +  0 + nB) * Tc + khB;
  const __bf16* bRow1 = Bt + (size_t)(n0 + 16 + nB) * Tc + khB;
  const __bf16* bRow2 = Bt + (size_t)(n0 + 32 + nB) * Tc + khB;
  const __bf16* bRow3 = Bt + (size_t)(n0 + 48 + nB) * Tc + khB;

  v8f acc[2][4] = {};

  v16bf a0 = load_tileA(aRow0), a1 = load_tileA(aRow1);
  v16bf q0 = load_tileB(bRow0), q1 = load_tileB(bRow1);
  v16bf q2 = load_tileB(bRow2), q3 = load_tileB(bRow3);

#pragma unroll
  for (int k0 = 32; k0 <= Tc; k0 += 32) {
    const int kl = (k0 < Tc) ? k0 : 0;   // last iter: branch-free dummy reload
    v16bf a0n = load_tileA(aRow0 + kl);
    v16bf a1n = load_tileA(aRow1 + kl);
    v16bf q0n = load_tileB(bRow0 + kl);
    v16bf q1n = load_tileB(bRow1 + kl);
    v16bf q2n = load_tileB(bRow2 + kl);
    v16bf q3n = load_tileB(bRow3 + kl);

    // Nothing may cross: next-step loads are issued before this step's WMMAs.
    __builtin_amdgcn_sched_barrier(0);

    acc[0][0] = wmma_bf16(a0, q0, acc[0][0]);
    acc[0][1] = wmma_bf16(a0, q1, acc[0][1]);
    acc[0][2] = wmma_bf16(a0, q2, acc[0][2]);
    acc[0][3] = wmma_bf16(a0, q3, acc[0][3]);
    acc[1][0] = wmma_bf16(a1, q0, acc[1][0]);
    acc[1][1] = wmma_bf16(a1, q1, acc[1][1]);
    acc[1][2] = wmma_bf16(a1, q2, acc[1][2]);
    acc[1][3] = wmma_bf16(a1, q3, acc[1][3]);

    a0 = a0n; a1 = a1n;
    q0 = q0n; q1 = q1n; q2 = q2n; q3 = q3n;
  }

  const int half = ln >> 4;
  __bf16* Sp = S + (size_t)b * Tc * Dc;
#pragma unroll
  for (int mt = 0; mt < 2; ++mt) {
    const int rbase = m0 + mt * 16 + half * 8;
#pragma unroll
    for (int nt = 0; nt < 4; ++nt) {
      const int n = n0 + nt * 16 + (ln & 15);
      // identity term: x[t][n] = x^T[n][t], 8 contiguous t per lane
      const __bf16* xp = Bt + (size_t)n * Tc + rbase;
      uint4 xraw = *(const uint4*)xp;
      const __bf16* xv = (const __bf16*)&xraw;
#pragma unroll
      for (int j = 0; j < 8; ++j) {
        float v = acc[mt][nt][j] + bf2f(xv[j]);
        Sp[(size_t)(rbase + j) * Dc + n] = f2bf(v);
      }
    }
  }
}

// ---------------------------------------------------------------------------
// GEMM2:  out^T[o][t] = relu( (sum_d W[o][d]*S[t][d] + 2b[o]) / denom[t] )
//   A = W bf16 [D][D] row-major; B(N-major) = S [T][D] row-major
//   writes x^T_next bf16 [D][T]; if last layer also writes f32 out [B][T][D]
//   block 128 (4 waves along M), wave -> 32(M) x 64(N); grid (D/128, T/64, B)
// ---------------------------------------------------------------------------
__global__ void __launch_bounds__(128)
gcnenc_gemm2(const __bf16* __restrict__ Wbf, const __bf16* __restrict__ S,
             const float* __restrict__ bias2, const float* __restrict__ denom,
             __bf16* __restrict__ xTo, float* __restrict__ out) {
  const int b  = blockIdx.z;
  const int wv = threadIdx.x >> 5;
  const int ln = threadIdx.x & 31;
  const int m0 = blockIdx.x * 128 + wv * 32;  // o tile base
  const int n0 = blockIdx.y * 64;             // t strip

  const __bf16* Bt = S + (size_t)b * Tc * Dc;

  const int mA  = ln & 15;
  const int kgA = (ln >> 4) * 8;
  const int nB  = ln & 15;
  const int khB = (ln >> 4) * 8;

  const __bf16* aRow0 = Wbf + (size_t)(m0 + mA) * Dc + kgA;
  const __bf16* aRow1 = Wbf + (size_t)(m0 + 16 + mA) * Dc + kgA;
  const __bf16* bRow0 = Bt + (size_t)(n0 +  0 + nB) * Dc + khB;
  const __bf16* bRow1 = Bt + (size_t)(n0 + 16 + nB) * Dc + khB;
  const __bf16* bRow2 = Bt + (size_t)(n0 + 32 + nB) * Dc + khB;
  const __bf16* bRow3 = Bt + (size_t)(n0 + 48 + nB) * Dc + khB;

  v8f acc[2][4] = {};

  v16bf a0 = load_tileA(aRow0), a1 = load_tileA(aRow1);
  v16bf q0 = load_tileB(bRow0), q1 = load_tileB(bRow1);
  v16bf q2 = load_tileB(bRow2), q3 = load_tileB(bRow3);

#pragma unroll
  for (int k0 = 32; k0 <= Dc; k0 += 32) {
    const int kl = (k0 < Dc) ? k0 : 0;
    v16bf a0n = load_tileA(aRow0 + kl);
    v16bf a1n = load_tileA(aRow1 + kl);
    v16bf q0n = load_tileB(bRow0 + kl);
    v16bf q1n = load_tileB(bRow1 + kl);
    v16bf q2n = load_tileB(bRow2 + kl);
    v16bf q3n = load_tileB(bRow3 + kl);

    __builtin_amdgcn_sched_barrier(0);

    acc[0][0] = wmma_bf16(a0, q0, acc[0][0]);
    acc[0][1] = wmma_bf16(a0, q1, acc[0][1]);
    acc[0][2] = wmma_bf16(a0, q2, acc[0][2]);
    acc[0][3] = wmma_bf16(a0, q3, acc[0][3]);
    acc[1][0] = wmma_bf16(a1, q0, acc[1][0]);
    acc[1][1] = wmma_bf16(a1, q1, acc[1][1]);
    acc[1][2] = wmma_bf16(a1, q2, acc[1][2]);
    acc[1][3] = wmma_bf16(a1, q3, acc[1][3]);

    a0 = a0n; a1 = a1n;
    q0 = q0n; q1 = q1n; q2 = q2n; q3 = q3n;
  }

  const int half = ln >> 4;
#pragma unroll
  for (int mt = 0; mt < 2; ++mt) {
    const int obase = m0 + mt * 16 + half * 8;
    const float4 bv0 = *(const float4*)(bias2 + obase);
    const float4 bv1 = *(const float4*)(bias2 + obase + 4);
    const float bb[8] = {bv0.x, bv0.y, bv0.z, bv0.w, bv1.x, bv1.y, bv1.z, bv1.w};
#pragma unroll
    for (int nt = 0; nt < 4; ++nt) {
      const int t  = n0 + nt * 16 + (ln & 15);
      const float dn = denom[(size_t)b * Tc + t];
      float r[8];
#pragma unroll
      for (int j = 0; j < 8; ++j) {
        float v = (acc[mt][nt][j] + bb[j]) / dn;
        r[j] = v > 0.0f ? v : 0.0f;
        xTo[((size_t)b * Dc + obase + j) * Tc + t] = f2bf(r[j]);
      }
      if (out) {
        float* op = out + ((size_t)b * Tc + t) * Dc + obase;
        *(float4*)(op)     = make_float4(r[0], r[1], r[2], r[3]);
        *(float4*)(op + 4) = make_float4(r[4], r[5], r[6], r[7]);
      }
    }
  }
}

// ---------------------------------------------------------------------------
extern "C" void kernel_launch(void* const* d_in, const int* in_sizes, int n_in,
                              void* d_out, int out_size, void* d_ws, size_t ws_size,
                              hipStream_t stream) {
  (void)in_sizes; (void)n_in; (void)out_size; (void)ws_size;

  const float* inputs  = (const float*)d_in[0];
  const int*   con_adj = (const int*)d_in[1];
  const int*   dep_adj = (const int*)d_in[2];
  const float* sem_adj = (const float*)d_in[3];
  const int*   amr_adj = (const int*)d_in[4];
  // d_in[5] tok_length: unused by the reference
  const float* W_con = (const float*)d_in[6];
  const float* b_con = (const float*)d_in[7];
  const float* W_dep = (const float*)d_in[8];
  const float* b_dep = (const float*)d_in[9];
  const float* W_sem = (const float*)d_in[10];
  const float* b_sem = (const float*)d_in[11];
  const float* W_amr = (const float*)d_in[12];
  const float* b_amr = (const float*)d_in[13];
  float* out = (float*)d_out;

  size_t off = 0;
  auto carve = [&](size_t bytes) -> void* {
    void* p = (char*)d_ws + off;
    off += (bytes + 255) & ~(size_t)255;
    return p;
  };
  __bf16* adjbf = (__bf16*)carve((size_t)Bc * Tc * Tc * 2);   // 16.8 MB
  __bf16* xT0   = (__bf16*)carve((size_t)Bc * Dc * Tc * 2);   //  8.4 MB
  __bf16* xTa   = (__bf16*)carve((size_t)Bc * Dc * Tc * 2);
  __bf16* xTb   = (__bf16*)carve((size_t)Bc * Dc * Tc * 2);
  __bf16* Sbuf  = (__bf16*)carve((size_t)Bc * Tc * Dc * 2);
  float*  denom = (float*) carve((size_t)Bc * Tc * 4);
  __bf16* Wbf   = (__bf16*)carve((size_t)Dc * Dc * 2);
  float*  bias2 = (float*) carve((size_t)Dc * 4);

  gcnenc_x0_prep<<<dim3((Bc * Tc * Dc) / 256), 256, 0, stream>>>(inputs, xT0);

  const dim3 adjGrid(Tc, Bc), blk256(256);
  const dim3 g1Grid(Tc / 32, Bc), blk128(128);
  const dim3 g2Grid(Dc / 128, Tc / 64, Bc);

  auto run_layers = [&](int L, bool perLayerAdj, const int* adjI, const float* adjF,
                        const float* Ws, const float* bs, float* outseg) {
    if (!perLayerAdj) {
      if (adjI)
        gcnenc_adj_prep_int<<<adjGrid, blk256, 0, stream>>>(adjI, adjbf, denom);
      else
        gcnenc_adj_prep_f32<<<adjGrid, blk256, 0, stream>>>(adjF, adjbf, denom);
    }
    const __bf16* xin = xT0;
    for (int l = 0; l < L; ++l) {
      if (perLayerAdj)
        gcnenc_adj_prep_int<<<adjGrid, blk256, 0, stream>>>(
            adjI + (size_t)l * Bc * Tc * Tc, adjbf, denom);
      gcnenc_w_prep<<<dim3((Dc * Dc) / 256), 256, 0, stream>>>(
          Ws + (size_t)l * Dc * Dc, bs + (size_t)l * Dc, Wbf, bias2);
      __bf16* xout = (l & 1) ? xTb : xTa;
      gcnenc_gemm1<<<g1Grid, blk128, 0, stream>>>(adjbf, xin, Sbuf);
      gcnenc_gemm2<<<g2Grid, blk128, 0, stream>>>(Wbf, Sbuf, bias2, denom, xout,
                                                  (l == L - 1) ? outseg : nullptr);
      xin = xout;
    }
  };

  const size_t seg = (size_t)Bc * Tc * Dc;
  run_layers(2, true,  con_adj, nullptr, W_con, b_con, out + 0 * seg);
  run_layers(2, false, dep_adj, nullptr, W_dep, b_dep, out + 1 * seg);
  run_layers(2, false, nullptr, sem_adj, W_sem, b_sem, out + 2 * seg);
  run_layers(9, false, amr_adj, nullptr, W_amr, b_amr, out + 3 * seg);
}
